// SelfTensProds_6167573037368
// MI455X (gfx1250) — matrix operand order repaired and checked by reading, not compiled, not run on hardware
//
#include <hip/hip_runtime.h>

// ---------------------------------------------------------------------------
// MI455X (gfx1250, wave32) implementation of SelfTensProds.
// All heavy contractions run on v_wmma_f32_16x16x32_f16 (f16 in, f32 acc).
// B operands (weights + P tensors) are pre-swizzled into WMMA fragment layout
// once per call; hot loops do 32B contiguous loads (L2-resident, 192MB).
// L/R tile staging in the tensor-product kernel uses gfx1250 async
// global->LDS copies (ASYNCcnt) when the toolchain exposes the builtins.
// ---------------------------------------------------------------------------

typedef _Float16 v16h __attribute__((ext_vector_type(16)));
typedef _Float16 v8h  __attribute__((ext_vector_type(8)));
typedef float    v8f  __attribute__((ext_vector_type(8)));

union V16 { v16h v; v8h h[2]; };

#if __has_builtin(__builtin_amdgcn_global_load_async_to_lds_b32) && \
    __has_builtin(__builtin_amdgcn_s_wait_asynccnt)
#define HAVE_ASYNC_LDS 1
#define AS1 __attribute__((address_space(1)))
#define AS3 __attribute__((address_space(3)))
#endif

// Gather a 16x32 f16 A-fragment from an LDS tile laid out [m][K] row-major.
// ISA 16-bit A layout: lane<16 holds row m=lane, K {0..7,16..23}; lane>=16
// holds row m=lane-16, K {8..15,24..31}.  Two 16B ds loads per fragment.
__device__ inline v16h load_a_frag(const _Float16* Atile, int Kstride, int kk, int lane) {
  const int m = lane & 15;
  const _Float16* row = Atile + m * Kstride + kk * 32 + ((lane & 16) ? 8 : 0);
  V16 a;
  a.h[0] = *(const v8h*)(row);
  a.h[1] = *(const v8h*)(row + 16);
  return a.v;
}

// ---------------------------------------------------------------------------
// B-fragment pre-swizzle: dst[((kstep*NT+ot)*32+lane)*16+e] = B(k,o) as f16,
// B(k,o) = (o<n_o1 ? s1[o*Kreal+k] : s2[(o-n_o1)*Kreal+k]), zero-padded k>=Kreal.
// Fragment layout: lane<16 -> K=kstep*32+{0..7,16..23}[e], col o=ot*16+lane;
//                  lane>=16 -> K=kstep*32+{8..15,24..31}[e], col o=ot*16+(lane-16).
// ---------------------------------------------------------------------------
__global__ void prep_frag_kernel(const float* s1, int n_o1, const float* s2,
                                 int Kreal, int Kpad, int NT, _Float16* dst, long total) {
  long idx = (long)blockIdx.x * blockDim.x + threadIdx.x;
  if (idx >= total) return;
  int e    = (int)(idx & 15);
  int lane = (int)((idx >> 4) & 31);
  long rest = idx >> 9;
  int ot    = (int)(rest % NT);
  int kstep = (int)(rest / NT);
  int k = kstep * 32 + (e & 7) + ((e & 8) ? 16 : 0) + ((lane & 16) ? 8 : 0);
  int o = ot * 16 + (lane & 15);
  float v = 0.0f;
  if (k < Kreal)
    v = (o < n_o1) ? s1[(long)o * Kreal + k] : s2[(long)(o - n_o1) * Kreal + k];
  dst[idx] = (_Float16)v;
}

__global__ void zero_kernel(float* p, long count) {
  long idx = (long)blockIdx.x * blockDim.x + threadIdx.x;
  if (idx < count) p[idx] = 0.0f;
}

// ---------------------------------------------------------------------------
// Generic strided GEMM: Out[n][c][o] = sum_k A[n][c][k] * B(k,o), 16-node tile
// per block, 8 waves; each wave owns output tiles ot = wv, wv+8, ...
// ---------------------------------------------------------------------------
struct GemmArgs {
  const float* A; long a_ns; long a_cs; long a_ks;
  int Kreal, Kpad, C, NT;
  const _Float16* B;
  float* Out; long o_ns; long o_cs;
};

__launch_bounds__(256)
__global__ void wmma_gemm_kernel(GemmArgs g) {
  __shared__ __align__(16) _Float16 Atile[16 * 256];
  const int tid  = threadIdx.x;
  const int lane = tid & 31;
  const int wv   = tid >> 5;
  const long n0  = (long)blockIdx.x * 16;
  const int K32  = g.Kpad >> 5;

  for (int comp = 0; comp < g.C; ++comp) {
    for (int idx = tid; idx < 16 * g.Kpad; idx += 256) {
      int m = idx / g.Kpad, k = idx - m * g.Kpad;
      float v = 0.0f;
      if (k < g.Kreal)
        v = g.A[(n0 + m) * g.a_ns + (long)comp * g.a_cs + (long)k * g.a_ks];
      Atile[idx] = (_Float16)v;
    }
    __syncthreads();
    for (int ot = wv; ot < g.NT; ot += 8) {
      v8f acc = {0.f, 0.f, 0.f, 0.f, 0.f, 0.f, 0.f, 0.f};
      for (int kk = 0; kk < K32; ++kk) {
        v16h a = load_a_frag(Atile, g.Kpad, kk, lane);
        const _Float16* bp = g.B + (((long)kk * g.NT + ot) * 32 + lane) * 16;
        int kpf = (kk + 1 < K32) ? kk + 1 : kk;   // clamped, branch-free prefetch
        __builtin_prefetch(g.B + (((long)kpf * g.NT + ot) * 32 + lane) * 16, 0, 1);
        v16h b = *(const v16h*)bp;
        acc = __builtin_amdgcn_wmma_f32_16x16x32_f16(false, a, false, b, (short)0, acc,
                                                     false, false);
      }
      // D layout: lane&15 = output column, VGPR j = node (base 0 / 8)
      int ocol  = ot * 16 + (lane & 15);
      int mbase = (lane & 16) ? 8 : 0;
      for (int j = 0; j < 8; ++j)
        g.Out[(n0 + mbase + j) * g.o_ns + (long)comp * g.o_cs + ocol] = acc[j];
    }
    __syncthreads();
  }
}

// ---------------------------------------------------------------------------
// Fused tensor-product branch:  y_c[n,o] += sum_k u_c[n,k] * P(k,o)
// u_c[n, l*dim_r+r] = sum_e w[e] * L[n,l,a[e]] * R[n,r,b[e]], entries sorted
// by c on host (cbeg ranges). 16-node tile per block; waves split K (grp) x
// output tiles (tgt), LDS f32 reduction combines K-partials.
// ---------------------------------------------------------------------------
struct TpArgs {
  const float* Lsrc; long l_ns; long l_cs; int l_off; int CL;
  const float* Rsrc; long r_ns; long r_cs; int r_off; int CR;
  int dim_r; int C; int NT; int gsplit;
  const _Float16* B;
  float* Out; long o_ns; long o_cs;
  int cbeg[10];               // entry ranges per output component (C<=9)
  int ea[28]; int eb[28]; float ew[28];
};

__launch_bounds__(256)
__global__ void tp_branch_kernel(TpArgs t) {
  __shared__ __align__(16) float    Ltile[16 * 16 * 9];   //  9216 B
  __shared__ __align__(16) float    Rtile[16 * 288];      // 18432 B
  __shared__ __align__(16) _Float16 U[16 * 1024];         // 32768 B
  __shared__ float accT[4 * 32 * 8];                      //  4096 B
  const int tid  = threadIdx.x;
  const int lane = tid & 31;
  const int wv   = tid >> 5;
  const long n0  = (long)blockIdx.x * 16;
  const int K    = t.dim_r * 16;
  const int K32  = K >> 5;

  // ---- stage L/R tiles: async global->LDS (ASYNCcnt) when available ----
  const int nL  = 16 * 16 * t.CL;
  const int per = t.dim_r * t.CR;
  const int nR  = 16 * per;
#ifdef HAVE_ASYNC_LDS
  for (int idx = tid; idx < nL; idx += 256) {
    int m = idx / (16 * t.CL);
    int rem = idx - m * 16 * t.CL;
    int l = rem / t.CL, a = rem - l * t.CL;
    const float* src = t.Lsrc + (n0 + m) * t.l_ns + (long)a * t.l_cs + t.l_off + l;
    __builtin_amdgcn_global_load_async_to_lds_b32(
        (AS1 int*)(unsigned long long)src, (AS3 int*)&Ltile[idx], 0, 0);
  }
  for (int idx = tid; idx < nR; idx += 256) {
    int m = idx / per;
    int rem = idx - m * per;
    int r = rem / t.CR, b = rem - r * t.CR;
    const float* src = t.Rsrc + (n0 + m) * t.r_ns + (long)b * t.r_cs + t.r_off + r;
    __builtin_amdgcn_global_load_async_to_lds_b32(
        (AS1 int*)(unsigned long long)src, (AS3 int*)&Rtile[idx], 0, 0);
  }
  __builtin_amdgcn_s_wait_asynccnt(0);
#else
  for (int idx = tid; idx < nL; idx += 256) {
    int m = idx / (16 * t.CL);
    int rem = idx - m * 16 * t.CL;
    int l = rem / t.CL, a = rem - l * t.CL;
    Ltile[idx] = t.Lsrc[(n0 + m) * t.l_ns + (long)a * t.l_cs + t.l_off + l];
  }
  for (int idx = tid; idx < nR; idx += 256) {
    int m = idx / per;
    int rem = idx - m * per;
    int r = rem / t.CR, b = rem - r * t.CR;
    Rtile[idx] = t.Rsrc[(n0 + m) * t.r_ns + (long)b * t.r_cs + t.r_off + r];
  }
#endif
  __syncthreads();

  const int tgt = wv % t.NT;   // output tile owned by this wave
  const int grp = wv / t.NT;   // K-split group

  for (int c = 0; c < t.C; ++c) {
    for (int idx = tid; idx < t.NT * 32 * 8; idx += 256) accT[idx] = 0.0f;
    // sparse pair product -> f16 A operand in LDS (only entries for this c)
    const int e0 = t.cbeg[c], e1 = t.cbeg[c + 1];
    for (int idx = tid; idx < 16 * K; idx += 256) {
      int m = idx / K, k = idx - m * K;
      int l = k / t.dim_r, r = k - l * t.dim_r;
      float s = 0.0f;
      for (int e = e0; e < e1; ++e)
        s += t.ew[e] * Ltile[(m * 16 + l) * t.CL + t.ea[e]] *
             Rtile[(m * t.dim_r + r) * t.CR + t.eb[e]];
      U[idx] = (_Float16)s;
    }
    __syncthreads();
    if (grp < t.gsplit) {
      v8f acc = {0.f, 0.f, 0.f, 0.f, 0.f, 0.f, 0.f, 0.f};
      for (int kk = grp; kk < K32; kk += t.gsplit) {
        v16h a = load_a_frag(U, K, kk, lane);
        const _Float16* bp = t.B + (((long)kk * t.NT + tgt) * 32 + lane) * 16;
        __builtin_prefetch(bp + t.gsplit * (long)t.NT * 512, 0, 1);
        v16h b = *(const v16h*)bp;
        acc = __builtin_amdgcn_wmma_f32_16x16x32_f16(false, a, false, b, (short)0, acc,
                                                     false, false);
      }
      float* ap = &accT[(tgt * 32 + lane) * 8];
      for (int j = 0; j < 8; ++j) atomicAdd(ap + j, acc[j]);  // ds_add_f32
    }
    __syncthreads();
    if (grp == 0) {
      int ocol  = tgt * 16 + (lane & 15);
      int mbase = (lane & 16) ? 8 : 0;
      const float* ap = &accT[(tgt * 32 + lane) * 8];
      for (int j = 0; j < 8; ++j) {
        long oi = (n0 + mbase + j) * t.o_ns + (long)c * t.o_cs + ocol;
        t.Out[oi] += ap[j];
      }
    }
    __syncthreads();
  }
}

// ---------------------------------------------------------------------------
// Residual + group norms (rank0 mean/var GN; rank1/2 RMS GN), one node/block.
// outV is [n][i][128] comp-major, outD is [n][ij][64]; transposed on write to
// match the reference (..., C, spatial) output layout.
// ---------------------------------------------------------------------------
__launch_bounds__(256)
__global__ void gn_kernel(const float* outA, const float* outV, const float* outD,
                          const float* x_a, const float* x_v, const float* x_d,
                          const float* g_a, const float* g_v, const float* g_d,
                          float* out, long N) {
  __shared__ float sA[256];
  __shared__ float sV[384];
  __shared__ float sD[576];
  __shared__ float stA_mu[8], stA_rs[8], stV_rs[8], stD_rs[8];
  const long n  = blockIdx.x;
  const int tid = threadIdx.x;

  sA[tid] = outA[n * 256 + tid] + x_a[n * 256 + tid];
  for (int idx = tid; idx < 384; idx += 256) {
    int c = idx / 3, i = idx - c * 3;
    sV[idx] = outV[n * 384 + (long)i * 128 + c] + x_v[n * 384 + idx];
  }
  for (int idx = tid; idx < 576; idx += 256) {
    int c = idx / 9, ij = idx - c * 9;
    sD[idx] = outD[n * 576 + (long)ij * 64 + c] + x_d[n * 576 + idx];
  }
  __syncthreads();
  if (tid < 8) {
    float s = 0.f, ss = 0.f;
    for (int j = 0; j < 32; ++j) { float v = sA[tid * 32 + j]; s += v; ss += v * v; }
    float mu = s * (1.0f / 32.0f);
    float var = ss * (1.0f / 32.0f) - mu * mu;
    stA_mu[tid] = mu;
    stA_rs[tid] = rsqrtf(var + 1e-5f);
  } else if (tid < 16) {
    int g = tid - 8;
    float ss = 0.f;
    for (int j = 0; j < 48; ++j) { float v = sV[g * 48 + j]; ss += v * v; }
    stV_rs[g] = rsqrtf(ss * (1.0f / 48.0f) + 1e-5f);
  } else if (tid < 24) {
    int g = tid - 16;
    float ss = 0.f;
    for (int j = 0; j < 72; ++j) { float v = sD[g * 72 + j]; ss += v * v; }
    stD_rs[g] = rsqrtf(ss * (1.0f / 72.0f) + 1e-5f);
  }
  __syncthreads();
  out[n * 256 + tid] = (sA[tid] - stA_mu[tid >> 5]) * stA_rs[tid >> 5] * g_a[tid];
  float* outv = out + N * 256;
  for (int idx = tid; idx < 384; idx += 256) {
    int c = idx / 3;
    outv[n * 384 + idx] = sV[idx] * stV_rs[c >> 4] * g_v[c];
  }
  float* outd = out + N * 256 + N * 384;
  for (int idx = tid; idx < 576; idx += 256) {
    int c = idx / 9;
    outd[n * 576 + idx] = sD[idx] * stD_rs[c >> 3] * g_d[c];
  }
}

// ---------------------------------------------------------------------------
// Host side
// ---------------------------------------------------------------------------
extern "C" void kernel_launch(void* const* d_in, const int* in_sizes, int n_in,
                              void* d_out, int out_size, void* d_ws, size_t ws_size,
                              hipStream_t stream) {
  (void)n_in; (void)out_size; (void)ws_size;
  const float* x_a  = (const float*)d_in[0];
  const float* x_v  = (const float*)d_in[1];
  const float* x_d  = (const float*)d_in[2];
  const float* P000 = (const float*)d_in[3];
  const float* P110 = (const float*)d_in[4];
  const float* P220 = (const float*)d_in[5];
  const float* P011 = (const float*)d_in[6];
  const float* P101 = (const float*)d_in[7];
  const float* P121 = (const float*)d_in[8];
  const float* P211 = (const float*)d_in[9];
  const float* P111 = (const float*)d_in[10];
  const float* P022 = (const float*)d_in[11];
  const float* P112 = (const float*)d_in[12];
  const float* P202 = (const float*)d_in[13];
  const float* P222 = (const float*)d_in[14];
  const float* P212 = (const float*)d_in[15];
  const float* W_left_a = (const float*)d_in[16];
  const float* W_left_v = (const float*)d_in[17];
  const float* W_left_d = (const float*)d_in[18];
  const float* W_a_in   = (const float*)d_in[19];
  const float* W_a_out  = (const float*)d_in[20];
  const float* W_v_in   = (const float*)d_in[21];
  const float* W_v_out  = (const float*)d_in[22];
  const float* W_d_in   = (const float*)d_in[23];
  const float* W_d_out  = (const float*)d_in[24];
  const float* g_a = (const float*)d_in[25];
  const float* g_v = (const float*)d_in[26];
  const float* g_d = (const float*)d_in[27];

  const long N = in_sizes[0] / 256;     // 16384 nodes
  const unsigned ntiles = (unsigned)(N / 16);

  // Workspace layout (f32): proj region is reused for out region after TP.
  float* wsf   = (float*)d_ws;
  float* projA = wsf;                   // [N][112]
  float* projV = projA + N * 112;       // [N][3][128]
  float* projD = projV + N * 384;       // [N][9][112]
  float* y_a   = wsf + N * 1504;        // [N][64]
  float* y_v   = y_a + N * 64;          // [N][3][48]
  float* y_d   = y_v + N * 144;         // [N][9][32]
  float* outA  = wsf;                   // alias over proj (safe: stage3 after TP)
  float* outV  = outA + N * 256;        // [N][3][128]
  float* outD  = outV + N * 384;        // [N][9][64]
  _Float16* frag = (_Float16*)(wsf + N * 2000);
  long fo = 0;

  // zero the y accumulators (TP branches add into them)
  {
    long cnt = N * 496;
    zero_kernel<<<dim3((unsigned)((cnt + 255) / 256)), dim3(256), 0, stream>>>(y_a, cnt);
  }

  auto prep = [&](const float* s1, int no1, const float* s2, int Kreal, int Kpad, int NT) {
    _Float16* dst = frag + fo;
    long total = (long)Kpad * NT * 16;
    fo += total;
    prep_frag_kernel<<<dim3((unsigned)((total + 255) / 256)), dim3(256), 0, stream>>>(
        s1, no1, s2, Kreal, Kpad, NT, dst, total);
    return dst;
  };

  _Float16* fWcatA = prep(W_left_a, 48, W_a_in, 256, 256, 7);   // 48 lefts + 64 mid
  _Float16* fWcatV = prep(W_left_v, 80, W_v_in, 128, 128, 8);   // 80 lefts + 48 mid
  _Float16* fWcatD = prep(W_left_d, 80, W_d_in, 64, 64, 7);     // 80 lefts + 32 mid
  _Float16* fWoutA = prep(W_a_out, 256, nullptr, 64, 64, 16);
  _Float16* fWoutV = prep(W_v_out, 128, nullptr, 48, 64, 8);
  _Float16* fWoutD = prep(W_d_out, 64, nullptr, 32, 32, 4);
  _Float16* fP000 = prep(P000, 64, nullptr, 1024, 1024, 4);
  _Float16* fP110 = prep(P110, 64, nullptr, 768, 768, 4);
  _Float16* fP220 = prep(P220, 64, nullptr, 512, 512, 4);
  _Float16* fP011 = prep(P011, 48, nullptr, 768, 768, 3);
  _Float16* fP101 = prep(P101, 48, nullptr, 1024, 1024, 3);
  _Float16* fP121 = prep(P121, 48, nullptr, 512, 512, 3);
  _Float16* fP211 = prep(P211, 48, nullptr, 768, 768, 3);
  _Float16* fP111 = prep(P111, 48, nullptr, 768, 768, 3);
  _Float16* fP022 = prep(P022, 32, nullptr, 512, 512, 2);
  _Float16* fP112 = prep(P112, 32, nullptr, 768, 768, 2);
  _Float16* fP202 = prep(P202, 32, nullptr, 1024, 1024, 2);
  _Float16* fP222 = prep(P222, 32, nullptr, 512, 512, 2);
  _Float16* fP212 = prep(P212, 32, nullptr, 768, 768, 2);

  auto gemm = [&](const float* A, long ans, long acs, long aks, int Kreal, int Kpad,
                  int C, int NT, const _Float16* B, float* Out, long ons, long ocs) {
    GemmArgs g;
    g.A = A; g.a_ns = ans; g.a_cs = acs; g.a_ks = aks;
    g.Kreal = Kreal; g.Kpad = Kpad; g.C = C; g.NT = NT;
    g.B = B; g.Out = Out; g.o_ns = ons; g.o_cs = ocs;
    wmma_gemm_kernel<<<dim3(ntiles), dim3(256), 0, stream>>>(g);
  };

  // Stage 1: left + mid projections
  gemm(x_a, 256, 0, 1, 256, 256, 1, 7, fWcatA, projA, 112, 0);
  gemm(x_v, 384, 1, 3, 128, 128, 3, 8, fWcatV, projV, 384, 128);
  gemm(x_d, 576, 1, 9, 64, 64, 9, 7, fWcatD, projD, 1008, 112);

  // Stage 2: 13 tensor-product branches
  struct Ent { int c, a, b; float w; };
  static const int epsnz[3][2][3] = {
      {{1, 2, 1}, {2, 1, -1}}, {{2, 0, 1}, {0, 2, -1}}, {{0, 1, 1}, {1, 0, -1}}};
  auto tpl = [&](const float* Ls, long lns, long lcs, int loff, int CL,
                 const float* Rs, long rns, long rcs, int roff, int CR,
                 int dim_r, int C, int NT, const _Float16* B,
                 float* Out, long ons, long ocs, const Ent* ents, int nnz) {
    TpArgs t{};
    t.Lsrc = Ls; t.l_ns = lns; t.l_cs = lcs; t.l_off = loff; t.CL = CL;
    t.Rsrc = Rs; t.r_ns = rns; t.r_cs = rcs; t.r_off = roff; t.CR = CR;
    t.dim_r = dim_r; t.C = C; t.NT = NT;
    t.gsplit = (NT == 2) ? 4 : 2;
    t.B = B; t.Out = Out; t.o_ns = ons; t.o_cs = ocs;
    // counting-sort entries by output component -> contiguous cbeg ranges
    int pos = 0;
    for (int c = 0; c < C; ++c) {
      t.cbeg[c] = pos;
      for (int e = 0; e < nnz; ++e) {
        if (ents[e].c == c) {
          t.ea[pos] = ents[e].a; t.eb[pos] = ents[e].b; t.ew[pos] = ents[e].w;
          ++pos;
        }
      }
    }
    t.cbeg[C] = pos;
    tp_branch_kernel<<<dim3(ntiles), dim3(256), 0, stream>>>(t);
  };

  { Ent e[28]; int n = 0; e[n++] = {0, 0, 0, 1.f};                 // 000
    tpl(projA, 112, 0, 0, 1, projA, 112, 0, 48, 1, 64, 1, 4, fP000, y_a, 64, 0, e, n); }
  { Ent e[28]; int n = 0; for (int i = 0; i < 3; i++) e[n++] = {0, i, i, 1.f};   // 110
    tpl(projV, 384, 128, 0, 3, projV, 384, 128, 80, 3, 48, 1, 4, fP110, y_a, 64, 0, e, n); }
  { Ent e[28]; int n = 0; for (int t9 = 0; t9 < 9; t9++) e[n++] = {0, t9, t9, 1.f}; // 220
    tpl(projD, 1008, 112, 0, 9, projD, 1008, 112, 80, 9, 32, 1, 4, fP220, y_a, 64, 0, e, n); }
  { Ent e[28]; int n = 0; for (int i = 0; i < 3; i++) e[n++] = {i, 0, i, 1.f};   // 011
    tpl(projA, 112, 0, 16, 1, projV, 384, 128, 80, 3, 48, 3, 3, fP011, y_v, 144, 48, e, n); }
  { Ent e[28]; int n = 0; for (int i = 0; i < 3; i++) e[n++] = {i, i, 0, 1.f};   // 101
    tpl(projV, 384, 128, 16, 3, projA, 112, 0, 48, 1, 64, 3, 3, fP101, y_v, 144, 48, e, n); }
  { Ent e[28]; int n = 0;                                                        // 121
    for (int i = 0; i < 3; i++) for (int j = 0; j < 3; j++) e[n++] = {i, j, i * 3 + j, 1.f};
    tpl(projV, 384, 128, 32, 3, projD, 1008, 112, 80, 9, 32, 3, 3, fP121, y_v, 144, 48, e, n); }
  { Ent e[28]; int n = 0;                                                        // 211
    for (int i = 0; i < 3; i++) for (int j = 0; j < 3; j++) e[n++] = {i, i * 3 + j, j, 1.f};
    tpl(projD, 1008, 112, 16, 9, projV, 384, 128, 80, 3, 48, 3, 3, fP211, y_v, 144, 48, e, n); }
  { Ent e[28]; int n = 0;                                                        // 111 (cross)
    for (int i = 0; i < 3; i++) for (int p = 0; p < 2; p++)
      e[n++] = {i, epsnz[i][p][0], epsnz[i][p][1], (float)epsnz[i][p][2]};
    tpl(projV, 384, 128, 48, 3, projV, 384, 128, 80, 3, 48, 3, 3, fP111, y_v, 144, 48, e, n); }
  { Ent e[28]; int n = 0; for (int t9 = 0; t9 < 9; t9++) e[n++] = {t9, 0, t9, 1.f}; // 022
    tpl(projA, 112, 0, 32, 1, projD, 1008, 112, 80, 9, 32, 9, 2, fP022, y_d, 288, 32, e, n); }
  { Ent e[28]; int n = 0; for (int t9 = 0; t9 < 9; t9++) e[n++] = {t9, t9, 0, 1.f}; // 202
    tpl(projD, 1008, 112, 32, 9, projA, 112, 0, 48, 1, 64, 9, 2, fP202, y_d, 288, 32, e, n); }
  { Ent e[28]; int n = 0;                                                        // 112 (outer)
    for (int i = 0; i < 3; i++) for (int j = 0; j < 3; j++) e[n++] = {i * 3 + j, i, j, 1.f};
    tpl(projV, 384, 128, 64, 3, projV, 384, 128, 80, 3, 48, 9, 2, fP112, y_d, 288, 32, e, n); }
  { Ent e[28]; int n = 0;                                                        // 222 (matmul)
    for (int i = 0; i < 3; i++) for (int j = 0; j < 3; j++) for (int k = 0; k < 3; k++)
      e[n++] = {i * 3 + j, i * 3 + k, k * 3 + j, 1.f};
    tpl(projD, 1008, 112, 48, 9, projD, 1008, 112, 80, 9, 32, 9, 2, fP222, y_d, 288, 32, e, n); }
  { Ent e[28]; int n = 0;                                                        // 212 (mat x cross)
    for (int i = 0; i < 3; i++) for (int j = 0; j < 3; j++) for (int p = 0; p < 2; p++)
      e[n++] = {i * 3 + j, i * 3 + epsnz[j][p][0], epsnz[j][p][1], (float)epsnz[j][p][2]};
    tpl(projD, 1008, 112, 64, 9, projV, 384, 128, 80, 3, 48, 9, 2, fP212, y_d, 288, 32, e, n); }

  // Stage 3: output projections
  gemm(y_a, 64, 0, 1, 64, 64, 1, 16, fWoutA, outA, 256, 0);
  gemm(y_v, 144, 48, 1, 48, 64, 3, 8, fWoutV, outV, 384, 128);
  gemm(y_d, 288, 32, 1, 32, 32, 9, 4, fWoutD, outD, 576, 64);

  // Stage 4: residual + group norms -> d_out
  gn_kernel<<<dim3((unsigned)N), dim3(256), 0, stream>>>(
      outA, outV, outD, x_a, x_v, x_d, g_a, g_v, g_d, (float*)d_out, N);
}
